// Attention_57990648431179
// MI455X (gfx1250) — matrix operand order repaired and checked by reading, not compiled, not run on hardware
//
#include <hip/hip_runtime.h>

// ---------------------------------------------------------------------------
// Problem constants (match the reference)
// ---------------------------------------------------------------------------
#define BB 16
#define NN 1024
#define DD 512
#define HH 8
#define HD 64              // head dim
#define MTOK (BB * NN)     // 16384 tokens

// ---------------------------------------------------------------------------
// Types / helpers
// ---------------------------------------------------------------------------
typedef __attribute__((ext_vector_type(16))) __bf16        v16bf;
typedef __attribute__((ext_vector_type(8)))  float         v8f;
typedef __attribute__((ext_vector_type(4)))  unsigned int  u32x4;

union BF16Frag {
    v16bf          v;      // 16 bf16 = 8 VGPRs (WMMA A/B operand, wave32)
    u32x4          q[2];   // two 16-byte chunks
    unsigned short s[16];
};

__device__ __forceinline__ unsigned short f2bf(float f) {
    unsigned int u = __builtin_bit_cast(unsigned int, f);
    u += 0x7FFFu + ((u >> 16) & 1u);          // round-to-nearest-even
    return (unsigned short)(u >> 16);
}

// WMMA D = A*B + C   (f32 accumulate, bf16 inputs, 16x16x32)
__device__ __forceinline__ v8f wmma_bf16(const BF16Frag& a, const BF16Frag& b, v8f c) {
    return __builtin_amdgcn_wmma_f32_16x16x32_bf16(
        /*neg_a=*/false, a.v, /*neg_b=*/false, b.v,
        /*c_mod=*/(short)0, c, /*reuse_a=*/false, /*reuse_b=*/false);
}

// A-operand (16 rows x 32 K, bf16, row-major source, row stride in elements).
// ISA layout: lanes 0-15 row M=lane, K 0..7 (v0-3) and 16..23 (v4-7);
//             lanes 16-31 row M=lane-16, K 8..15 and 24..31.
__device__ __forceinline__ BF16Frag load_a_frag(const unsigned short* base,
                                                int row0, int rowstride, int k0) {
    const int lane = threadIdx.x & 31;
    const int hsel = lane >> 4;
    const unsigned short* p =
        base + (size_t)(row0 + (lane & 15)) * rowstride + k0 + hsel * 8;
    BF16Frag f;
    f.q[0] = *(const u32x4*)(p);
    f.q[1] = *(const u32x4*)(p + 16);
    return f;
}

// B-operand (32 K x 16 cols) where B(k, n) = SRC[n][k] with SRC row-major
// (row stride in elements).  ISA layout: lanes 0-15 column N=lane, K 0..7 and 16..23;
// lanes 16-31 column N=lane-16, K 8..15 and 24..31 (same order as A).
__device__ __forceinline__ BF16Frag load_bT_frag(const unsigned short* base,
                                                 int row0, int rowstride, int k0) {
    const int lane = threadIdx.x & 31;
    const int hsel = lane >> 4;
    const unsigned short* p =
        base + (size_t)(row0 + (lane & 15)) * rowstride + k0 + hsel * 8;
    BF16Frag f;
    f.q[0] = *(const u32x4*)(p);
    f.q[1] = *(const u32x4*)(p + 16);
    return f;
}

// ---------------------------------------------------------------------------
// CDNA5 async copy: global -> LDS, 16 bytes per lane, tracked by ASYNCcnt.
// VGLOBAL "GV" form: vdst = LDS byte address, vaddr = 64-bit global address.
// Generic pointers to __shared__ carry the LDS offset in their low 32 bits
// (flat-aperture rule).
// ---------------------------------------------------------------------------
__device__ __forceinline__ void async_load_b128(const void* gaddr, void* lds) {
    unsigned ldsa = (unsigned)(uintptr_t)lds;
    unsigned long long ga = (unsigned long long)(uintptr_t)gaddr;
    asm volatile("global_load_async_to_lds_b128 %0, %1, off"
                 :: "v"(ldsa), "v"(ga)
                 : "memory");
}
template <int N>
__device__ __forceinline__ void wait_async() {
    asm volatile("s_wait_asynccnt %0" :: "i"(N) : "memory");
}

// ---------------------------------------------------------------------------
// f32 -> bf16 conversion pass
// ---------------------------------------------------------------------------
__global__ void __launch_bounds__(256) cvt_bf16_kernel(const float* __restrict__ src,
                                                       unsigned short* __restrict__ dst,
                                                       int n) {
    int i = blockIdx.x * 256 + threadIdx.x;
    if (i < n) dst[i] = f2bf(src[i]);
}

// ---------------------------------------------------------------------------
// GEMM:  Y = A @ W^T + bias      A:[MTOK,512] bf16, W:[512,512] bf16 row-major
// Each wave computes a 16(M) x 64(N) tile; K loop in steps of 32.
//   mode 0: Y bf16 [MTOK,512]                     (Q, K projections)
//   mode 1: Y bf16 transposed per (b,h): [B*H*64, N]  (V -> V^T for PV matmul)
//   mode 2: Y f32  [MTOK,512]                     (output projection -> d_out)
// ---------------------------------------------------------------------------
__global__ void __launch_bounds__(256) gemm_wmma_kernel(
    const unsigned short* __restrict__ A,
    const unsigned short* __restrict__ W,
    const float* __restrict__ bias,
    void* __restrict__ Y, int mode)
{
    const int lane = threadIdx.x & 31;
    const int hsel = lane >> 4;
    const int col  = lane & 15;
    const int wid  = blockIdx.x * 8 + (threadIdx.x >> 5);
    const int mt   = wid >> 3;           // 0..1023  (M tile)
    const int nt   = wid & 7;            // 0..7     (N tile of 64)
    const int row0 = mt * 16;
    const int col0 = nt * 64;

    v8f acc0 = (v8f)0.0f, acc1 = (v8f)0.0f, acc2 = (v8f)0.0f, acc3 = (v8f)0.0f;

#pragma unroll 1
    for (int kk = 0; kk < DD; kk += 32) {
        BF16Frag a = load_a_frag(A, row0, DD, kk);
        BF16Frag b0 = load_bT_frag(W, col0 +  0, DD, kk);
        BF16Frag b1 = load_bT_frag(W, col0 + 16, DD, kk);
        BF16Frag b2 = load_bT_frag(W, col0 + 32, DD, kk);
        BF16Frag b3 = load_bT_frag(W, col0 + 48, DD, kk);
        acc0 = wmma_bf16(a, b0, acc0);
        acc1 = wmma_bf16(a, b1, acc1);
        acc2 = wmma_bf16(a, b2, acc2);
        acc3 = wmma_bf16(a, b3, acc3);
    }

    v8f acc[4] = {acc0, acc1, acc2, acc3};

    if (mode == 0) {                 // bf16, row-major [MTOK, 512]
        unsigned short* O = (unsigned short*)Y;
#pragma unroll
        for (int c = 0; c < 4; ++c) {
            const int n = col0 + c * 16 + col;
            const float bc = bias[n];
#pragma unroll
            for (int j = 0; j < 8; ++j) {
                const int r = row0 + j + hsel * 8;
                O[(size_t)r * DD + n] = f2bf(acc[c][j] + bc);
            }
        }
    } else if (mode == 1) {          // bf16, transposed per (b,h): [B*H*64, N]
        unsigned short* O = (unsigned short*)Y;
        const int tok0 = row0 + hsel * 8;       // 8 consecutive tokens per lane
        const int bidx = tok0 >> 10;
        const int n0   = tok0 & (NN - 1);
#pragma unroll
        for (int c = 0; c < 4; ++c) {
            const int f = col0 + c * 16 + col;  // output feature
            const float bc = bias[f];
            union { unsigned short s[8]; u32x4 q; } t;
#pragma unroll
            for (int j = 0; j < 8; ++j) t.s[j] = f2bf(acc[c][j] + bc);
            const int h  = f >> 6;
            const int dd = f & 63;
            *(u32x4*)(O + ((size_t)((bidx * HH + h) * HD + dd)) * NN + n0) = t.q;
        }
    } else {                         // f32, row-major [MTOK, 512] (final output)
        float* O = (float*)Y;
#pragma unroll
        for (int c = 0; c < 4; ++c) {
            const int n = col0 + c * 16 + col;
            const float bc = bias[n];
#pragma unroll
            for (int j = 0; j < 8; ++j) {
                const int r = row0 + j + hsel * 8;
                O[(size_t)r * DD + n] = acc[c][j] + bc;
            }
        }
    }
}

// ---------------------------------------------------------------------------
// Flash attention, workgroup-cooperative:
//   block = one (b,h) and 8 query tiles (8 waves x 16 queries).
//   K/V^T key-blocks (32 keys) are DMA'd into double-buffered LDS with
//   global_load_async_to_lds_b128 (ASYNCcnt); next block's DMA overlaps the
//   current block's WMMAs (wait asynccnt<=2: async loads complete in order,
//   each thread keeps exactly its 2 next-tile ops in flight).
//   energy = Q K^T / 8 ; online softmax (denominator unmasked, as in ref);
//   p *= mask[q,k] (post-softmax multiplicative mask); acc += P @ V via
//   LDS-transposed P (A-operand) and pre-transposed V^T (B-operand).
// ---------------------------------------------------------------------------
__global__ void __launch_bounds__(256) attn_wmma_kernel(
    const unsigned short* __restrict__ Q,    // [B*N, 512] bf16
    const unsigned short* __restrict__ K,    // [B*N, 512] bf16
    const unsigned short* __restrict__ Vt,   // [B*H*64, N] bf16 (transposed V)
    const float* __restrict__ mask,          // [N, N] f32
    unsigned short* __restrict__ O)          // [B*N, 512] bf16 head_out
{
    __shared__ unsigned short kbuf[2][32 * 64];   // 32 keys x 64 d   (4 KB each)
    __shared__ unsigned short vbuf[2][64 * 32];   // 64 d   x 32 keys (4 KB each)
    __shared__ unsigned short pbuf[8][16 * 32];   // per-wave 16x32 P tile (8 KB)

    const int tid  = threadIdx.x;
    const int wave = tid >> 5;
    const int lane = tid & 31;
    const int hsel = lane >> 4;
    const int col  = lane & 15;

    const int bh = blockIdx.x >> 3;            // 0..127 -> (b,h)
    const int qg = blockIdx.x & 7;             // query-tile group
    const int b  = bh >> 3;
    const int h  = bh & 7;
    const int q0 = (qg * 8 + wave) * 16;       // this wave's 16 queries

    const unsigned short* Qh = Q + (size_t)b * NN * DD + h * HD;
    const unsigned short* Kh = K + (size_t)b * NN * DD + h * HD;
    const unsigned short* Vh = Vt + (size_t)(b * HH + h) * HD * NN;
    unsigned short* pw = pbuf[wave];

    // cooperative DMA of one 32-key block of K (32x64) and V^T (64x32):
    // 256 threads x (16B K-chunk + 16B V-chunk) = 2 async ops per thread.
    const int krow = tid >> 3, kch = (tid & 7) * 8;   // K: 32 rows x 8 chunks
    const int vrow = tid >> 2, vch = (tid & 3) * 8;   // V: 64 rows x 4 chunks

    // Q tile is reused across the whole key loop: keep both K-halves resident.
    const BF16Frag aQlo = load_a_frag(Qh, q0, DD, 0);
    const BF16Frag aQhi = load_a_frag(Qh, q0, DD, 32);

    v8f acc[4];
    acc[0] = (v8f)0.0f; acc[1] = (v8f)0.0f; acc[2] = (v8f)0.0f; acc[3] = (v8f)0.0f;
    float mrow[8], lrow[8];
#pragma unroll
    for (int j = 0; j < 8; ++j) { mrow[j] = -1e30f; lrow[j] = 0.0f; }

    // prime the pipeline: stage key-block 0 into buffer 0
    async_load_b128(Kh + (size_t)krow * DD + kch,      &kbuf[0][krow * 64 + kch]);
    async_load_b128(Vh + (size_t)vrow * NN + vch,      &vbuf[0][vrow * 32 + vch]);

#pragma unroll 1
    for (int kb = 0; kb < NN; kb += 32) {
        const int cur = (kb >> 5) & 1;

        if (kb + 32 < NN) {       // stage next block into the other buffer
            async_load_b128(Kh + (size_t)(kb + 32 + krow) * DD + kch,
                            &kbuf[cur ^ 1][krow * 64 + kch]);
            async_load_b128(Vh + (size_t)vrow * NN + (kb + 32) + vch,
                            &vbuf[cur ^ 1][vrow * 32 + vch]);
            wait_async<2>();      // current block landed; next stays in flight
        } else {
            wait_async<0>();
        }
        __syncthreads();          // staged LDS visible to all 8 waves

        const unsigned short* kb_lds = kbuf[cur];
        const unsigned short* vb_lds = vbuf[cur];

        // ---- energies: S = Q K^T (16 queries x 32 keys), K-dim 64 = 2x32 ----
        v8f s0 = (v8f)0.0f, s1 = (v8f)0.0f;
        {
            BF16Frag bk;
            bk = load_bT_frag(kb_lds,  0, 64, 0);  s0 = wmma_bf16(aQlo, bk, s0);
            bk = load_bT_frag(kb_lds,  0, 64, 32); s0 = wmma_bf16(aQhi, bk, s0);
            bk = load_bT_frag(kb_lds, 16, 64, 0);  s1 = wmma_bf16(aQlo, bk, s1);
            bk = load_bT_frag(kb_lds, 16, 64, 32); s1 = wmma_bf16(aQhi, bk, s1);
        }

        // ---- online softmax + post-softmax mask, write P tile to LDS ----
        const float scl = 0.125f;                  // 1/sqrt(64)
#pragma unroll
        for (int j = 0; j < 8; ++j) {
            const float e0 = s0[j] * scl;
            const float e1 = s1[j] * scl;
            float t = fmaxf(e0, e1);               // row max over this key block
            t = fmaxf(t, __shfl_xor(t, 1));
            t = fmaxf(t, __shfl_xor(t, 2));
            t = fmaxf(t, __shfl_xor(t, 4));
            t = fmaxf(t, __shfl_xor(t, 8));        // stays within 16-lane row group
            const float nm = fmaxf(mrow[j], t);
            const float rs = __expf(mrow[j] - nm); // rescale old state
            mrow[j] = nm;
            lrow[j] *= rs;
            acc[0][j] *= rs; acc[1][j] *= rs; acc[2][j] *= rs; acc[3][j] *= rs;
            float p0 = __expf(e0 - nm);
            float p1 = __expf(e1 - nm);
            lrow[j] += p0 + p1;                    // denominator: UNmasked (ref)
            const int qrow = q0 + j + hsel * 8;
            p0 *= mask[(size_t)qrow * NN + kb + col];
            p1 *= mask[(size_t)qrow * NN + kb + 16 + col];
            const int prow = j + hsel * 8;         // C-layout row for this VGPR
            pw[prow * 32 + col]      = f2bf(p0);
            pw[prow * 32 + 16 + col] = f2bf(p1);
        }

        // ---- P (16x32) as A-operand from LDS; V^T rows as B-operand ----
        const BF16Frag ap = load_a_frag(pw, 0, 32, 0);
#pragma unroll
        for (int c = 0; c < 4; ++c) {
            BF16Frag bv = load_bT_frag(vb_lds, c * 16, 32, 0);
            acc[c] = wmma_bf16(ap, bv, acc[c]);
        }

        __syncthreads();          // all waves done reading before next overwrite
    }

    // ---- finalize: divide by full softmax denominator, store head_out ----
#pragma unroll
    for (int j = 0; j < 8; ++j) {
        float l = lrow[j];
        l += __shfl_xor(l, 1);
        l += __shfl_xor(l, 2);
        l += __shfl_xor(l, 4);
        l += __shfl_xor(l, 8);
        const float inv = 1.0f / l;
        const int r = q0 + j + hsel * 8;
        const size_t base = ((size_t)b * NN + r) * DD + h * HD;
#pragma unroll
        for (int c = 0; c < 4; ++c)
            O[base + c * 16 + col] = f2bf(acc[c][j] * inv);
    }
}

// ---------------------------------------------------------------------------
// Launch
// ---------------------------------------------------------------------------
extern "C" void kernel_launch(void* const* d_in, const int* in_sizes, int n_in,
                              void* d_out, int out_size, void* d_ws, size_t ws_size,
                              hipStream_t stream) {
    const float* x    = (const float*)d_in[0];
    const float* mask = (const float*)d_in[1];
    const float* Wq   = (const float*)d_in[2];
    const float* bq   = (const float*)d_in[3];
    const float* Wk   = (const float*)d_in[4];
    const float* bk   = (const float*)d_in[5];
    const float* Wv   = (const float*)d_in[6];
    const float* bv   = (const float*)d_in[7];
    const float* Wp   = (const float*)d_in[8];
    const float* bp   = (const float*)d_in[9];

    // workspace layout (bf16 buffers), all 16B aligned
    char* ws = (char*)d_ws;
    unsigned short* xb  = (unsigned short*)(ws);                         // 16 MB
    unsigned short* Wqb = (unsigned short*)(ws + (size_t)16 * 1024 * 1024);
    unsigned short* Wkb = Wqb + DD * DD;
    unsigned short* Wvb = Wkb + DD * DD;
    unsigned short* Wpb = Wvb + DD * DD;
    unsigned short* Qb  = Wpb + DD * DD;                                 // 16 MB
    unsigned short* Kb  = Qb + (size_t)MTOK * DD;
    unsigned short* Vtb = Kb + (size_t)MTOK * DD;                        // [B*H*64, N]
    unsigned short* Hb  = Vtb + (size_t)MTOK * DD;                       // head_out

    const int nx = MTOK * DD;      // 8,388,608
    const int nw = DD * DD;        // 262,144

    cvt_bf16_kernel<<<(nx + 255) / 256, 256, 0, stream>>>(x,  xb,  nx);
    cvt_bf16_kernel<<<(nw + 255) / 256, 256, 0, stream>>>(Wq, Wqb, nw);
    cvt_bf16_kernel<<<(nw + 255) / 256, 256, 0, stream>>>(Wk, Wkb, nw);
    cvt_bf16_kernel<<<(nw + 255) / 256, 256, 0, stream>>>(Wv, Wvb, nw);
    cvt_bf16_kernel<<<(nw + 255) / 256, 256, 0, stream>>>(Wp, Wpb, nw);

    // 8192 wave-tiles = 1024 blocks x 8 waves (wave32)
    dim3 gblk(1024), gthr(256);
    gemm_wmma_kernel<<<gblk, gthr, 0, stream>>>(xb, Wqb, bq, Qb,  0);
    gemm_wmma_kernel<<<gblk, gthr, 0, stream>>>(xb, Wkb, bk, Kb,  0);
    gemm_wmma_kernel<<<gblk, gthr, 0, stream>>>(xb, Wvb, bv, Vtb, 1);

    attn_wmma_kernel<<<gblk, gthr, 0, stream>>>(Qb, Kb, Vtb, mask, Hb);

    gemm_wmma_kernel<<<gblk, gthr, 0, stream>>>(Hb, Wpb, bp, d_out, 2);
}
